// Linear4bit_20675972563491
// MI455X (gfx1250) — matrix operand order repaired and checked by reading, not compiled, not run on hardware
//
#include <hip/hip_runtime.h>

#define IN_FEAT  4096
#define OUT_FEAT 11008
#define QCOLS    (OUT_FEAT / 8)   // 1376 packed int32 columns

typedef _Float16 v16h __attribute__((ext_vector_type(16)));
typedef _Float16 v8h  __attribute__((ext_vector_type(8)));
typedef _Float16 v4h  __attribute__((ext_vector_type(4)));
typedef float    v8f  __attribute__((ext_vector_type(8)));
typedef float    v4f  __attribute__((ext_vector_type(4)));

#define BM 128
#define BN 128
#define BK 64
#define KPAD 72   // halves: 144B row stride -> 16B aligned b128 reads, conflict-free banks
#define KTILES (IN_FEAT / BK)

__global__ __launch_bounds__(256, 2)
void w4a16_wmma_gemm(const float* __restrict__ x,
                     const int*   __restrict__ qweight,
                     const float* __restrict__ scales,
                     const int*   __restrict__ qzeros,
                     const float* __restrict__ bias,
                     float* __restrict__ out,
                     int M)
{
    __shared__ _Float16 lA[2][BM * KPAD];   // double-buffered A tile, f16, [m][k]
    __shared__ _Float16 lB[2][BN * KPAD];   // double-buffered B tile, f16, [n][k]
    __shared__ float    sS[BN];             // per-col scale
    __shared__ float    sZS[BN];            // per-col scale*zero

    const int tid  = threadIdx.x;
    const int lane = tid & 31;
    const int wave = tid >> 5;
    const int lrow = lane & 15;
    const int lhi  = lane >> 4;             // 0 or 1

    const int n0 = blockIdx.x * BN;
    const int m0 = blockIdx.y * BM;

    // wave tile: 32 (M) x 64 (N); waves arranged 4 (M) x 2 (N)
    const int wm0 = (wave & 3) * 32;
    const int wn0 = (wave >> 2) * 64;

    // A staging: 2 threads per row, 32 consecutive floats each
    const int arow = tid >> 1;
    const int acol = (tid & 1) * 32;
    // B staging: one 4(K)x8(N) micro-tile per thread
    const int bg  = tid & 15;   // packed column group: n = bg*8 .. bg*8+7
    const int bkc = tid >> 4;   // k chunk: k = bkc*4 .. bkc*4+3 (covers BK=64)

    // ---- register prefetch state ----
    v4f          pa[8];
    unsigned int pq[4];

    const float* srcA0 = x + (long)(m0 + arow) * IN_FEAT + acol;
    const int*   srcB0 = qweight + (long)(bkc * 4) * QCOLS + (n0 >> 3) + bg;

    auto prefetch = [&](int kk) {
        const float* srcA = srcA0 + kk;
#pragma unroll
        for (int c = 0; c < 8; ++c) pa[c] = *(const v4f*)(srcA + c * 4);
        const int* srcB = srcB0 + (long)kk * QCOLS;
#pragma unroll
        for (int i = 0; i < 4; ++i) pq[i] = (unsigned int)srcB[i * QCOLS];
    };

    prefetch(0);

    // ---- preload per-column scale / scale*zero into LDS ----
    if (tid < BN) {
        const int   n = n0 + tid;
        const float s = scales[n];
        const int   q = qzeros[n >> 3];
        const int   z = (q >> ((n & 7) * 4)) & 0xF;
        sS[tid]  = s;
        sZS[tid] = s * (float)z;
    }

    v8f acc[2][4];
    const v8f vzero = {0.f, 0.f, 0.f, 0.f, 0.f, 0.f, 0.f, 0.f};
#pragma unroll
    for (int i = 0; i < 2; ++i)
#pragma unroll
        for (int j = 0; j < 4; ++j) acc[i][j] = vzero;

    __syncthreads();   // sS/sZS visible before first dequant-store

    for (int t = 0; t < KTILES; ++t) {
        const int buf = t & 1;

        // ---- store prefetched A (f32 -> f16) ----
        {
            _Float16* dst = &lA[buf][arow * KPAD + acol];
#pragma unroll
            for (int c = 0; c < 4; ++c) {
                v8f f = __builtin_shufflevector(pa[2 * c], pa[2 * c + 1],
                                                0, 1, 2, 3, 4, 5, 6, 7);
                *(v8h*)(dst + c * 8) = __builtin_convertvector(f, v8h);
            }
        }

        // ---- store prefetched B (unpack + dequant -> f16, transposed [n][k]) ----
        {
#pragma unroll
            for (int j = 0; j < 8; ++j) {
                const int   nl = bg * 8 + j;
                const float s  = sS[nl];
                const float zs = sZS[nl];
                v4h h;
#pragma unroll
                for (int i = 0; i < 4; ++i) {
                    const float w = (float)((pq[i] >> (4 * j)) & 0xF) * s - zs;
                    h[i] = (_Float16)w;
                }
                *(v4h*)(&lB[buf][nl * KPAD + bkc * 4]) = h;
            }
        }

        __syncthreads();   // tile `buf` visible; also fences iter t-1 readers of buf^1

        // ---- prefetch next tile (overlaps compute below) ----
        if (t + 1 < KTILES) prefetch((t + 1) * BK);

        // ---- compute: two K=32 WMMA steps from LDS[buf] ----
#pragma unroll
        for (int ks = 0; ks < BK; ks += 32) {
            v16h afrag[2], bfrag[4];
#pragma unroll
            for (int mt = 0; mt < 2; ++mt) {
                // A 16x32 f16 layout: lane holds M=lane%16; K halves [b0,b0+8) and [16+b0,+8), b0=8*lhi
                const _Float16* p = &lA[buf][(wm0 + mt * 16 + lrow) * KPAD + ks + lhi * 8];
                v8h lo = *(const v8h*)p;
                v8h hi = *(const v8h*)(p + 16);
                afrag[mt] = __builtin_shufflevector(lo, hi, 0, 1, 2, 3, 4, 5, 6, 7,
                                                    8, 9, 10, 11, 12, 13, 14, 15);
            }
#pragma unroll
            for (int nt = 0; nt < 4; ++nt) {
                // B 32x16 f16 layout: lane holds N=lane%16; K halves [b0,b0+8) and [16+b0,+8), b0=8*lhi
                const _Float16* p = &lB[buf][(wn0 + nt * 16 + lrow) * KPAD + ks + lhi * 8];
                v8h lo = *(const v8h*)p;
                v8h hi = *(const v8h*)(p + 16);
                bfrag[nt] = __builtin_shufflevector(lo, hi, 0, 1, 2, 3, 4, 5, 6, 7,
                                                    8, 9, 10, 11, 12, 13, 14, 15);
            }
#pragma unroll
            for (int mt = 0; mt < 2; ++mt)
#pragma unroll
                for (int nt = 0; nt < 4; ++nt)
                    acc[mt][nt] = __builtin_amdgcn_wmma_f32_16x16x32_f16(
                        false, afrag[mt], false, bfrag[nt],
                        (short)0, acc[mt][nt], false, false);
        }
    }

    // ---- epilogue: C layout VGPR r -> M = r + 8*lhi, N = lane%16 ----
    // Non-temporal stores: out is write-once (361 MB); keep x (134 MB) resident in L2.
#pragma unroll
    for (int nt = 0; nt < 4; ++nt) {
        const int   n = n0 + wn0 + nt * 16 + lrow;
        const float b = bias[n];
#pragma unroll
        for (int mt = 0; mt < 2; ++mt) {
            const int mbase = m0 + wm0 + mt * 16 + lhi * 8;
#pragma unroll
            for (int r = 0; r < 8; ++r) {
                __builtin_nontemporal_store(acc[mt][nt][r] + b,
                                            &out[(long)(mbase + r) * OUT_FEAT + n]);
            }
        }
    }
}

extern "C" void kernel_launch(void* const* d_in, const int* in_sizes, int n_in,
                              void* d_out, int out_size, void* d_ws, size_t ws_size,
                              hipStream_t stream) {
    const float* x       = (const float*)d_in[0];
    const int*   qweight = (const int*)d_in[1];
    const float* scales  = (const float*)d_in[2];
    const int*   qzeros  = (const int*)d_in[3];
    const float* bias    = (const float*)d_in[4];
    float*       out     = (float*)d_out;

    const int M = in_sizes[0] / IN_FEAT;   // 4*2048 = 8192 rows

    dim3 grid(OUT_FEAT / BN, M / BM);      // (86, 64)
    dim3 block(256);
    w4a16_wmma_gemm<<<grid, block, 0, stream>>>(x, qweight, scales, qzeros, bias, out, M);
}
